// HuffmanWeight_56745107915432
// MI455X (gfx1250) — matrix operand order repaired and checked by reading, not compiled, not run on hardware
//
#include <hip/hip_runtime.h>
#include <hip/hip_bf16.h>

// y[4,8192] = x[4,8192] . W^T + bias, W decoded from (exp<<8)|rem byte planes.
// HBM-bound: 512MB of int32 byte-plane traffic -> ~22us floor at 23.3 TB/s.
// Stream planes with non-temporal b128 loads, decode via v_perm_b32 into
// packed bf16, feed v_wmma_f32_16x16x32_bf16, split K over 8 waves/block,
// reduce through LDS, add bias, store bf16.

typedef __attribute__((ext_vector_type(16))) __bf16          v16bf;
typedef __attribute__((ext_vector_type(8)))  float           v8f;
typedef __attribute__((ext_vector_type(4)))  int             i4v;
typedef __attribute__((ext_vector_type(8)))  unsigned short  u16x8;

#define MDIM   8192
#define NDIM   4096
#define KDIM   8192
#define BATCH  4
#define WAVES  8
#define KCHUNK (KDIM / WAVES)   // 1024 K per wave (stays inside one split plane)
#define KITER  (KCHUNK / 32)    // 32 WMMA iterations per wave

// Build one packed-bf16 dword from two rem dwords + two exp dwords:
//   result bytes = { e_hi, r_hi, e_lo, r_lo }  (little-endian)
__device__ __forceinline__ unsigned int pack2(unsigned int r_lo, unsigned int r_hi,
                                              unsigned int e_lo, unsigned int e_hi) {
    unsigned int tr = __builtin_amdgcn_perm(r_hi, r_lo, 0x0C040C00u); // r0->b0, r1->b2
    unsigned int te = __builtin_amdgcn_perm(e_hi, e_lo, 0x040C000Cu); // e0->b1, e1->b3
    return tr | te;
}

__global__ __launch_bounds__(WAVES * 32)
void huff_gemv_wmma(const unsigned short* __restrict__ x,     // [4,8192] bf16 bits
                    const int*            __restrict__ rem,   // [2,8192,4096] low byte
                    const int*            __restrict__ expb,  // [2,8192,4096] high byte
                    const unsigned short* __restrict__ bias,  // [8192] bf16 bits
                    unsigned short*       __restrict__ out)   // [4,8192] bf16 bits
{
    __shared__ float red[WAVES][32][8];   // 8 KB: per-wave f32 partial D tiles

    const int tid  = threadIdx.x;
    const int wave = tid >> 5;
    const int lane = tid & 31;
    const int col  = lane & 15;   // B-matrix column == output m offset in tile
    const int half = lane >> 4;   // K-half selector (B: k 0-15 vs 16-31)

    const int m0 = blockIdx.x * 16;

    // This wave's K range; chunk boundaries align with the split_k planes.
    const int kw = wave * KCHUNK;          // global k start
    const int s  = (kw >= NDIM) ? 1 : 0;   // which byte plane
    const int n0 = kw - s * NDIM;          // n start inside plane

    const size_t plane = (size_t)s * MDIM * NDIM;
    const int* remRow = rem  + plane + (size_t)(m0 + col) * NDIM + n0 + half * 16;
    const int* expRow = expb + plane + (size_t)(m0 + col) * NDIM + n0 + half * 16;

    // A-operand source row. Rows >= BATCH are don't-care: they only feed
    // D rows we never store, so just clamp the address (no zeroing needed).
    const int brow = col;
    const unsigned short* xBase =
        x + (size_t)(brow < BATCH ? brow : 0) * KDIM + kw;

    v8f acc = {0.f, 0.f, 0.f, 0.f, 0.f, 0.f, 0.f, 0.f};

    #pragma unroll 2
    for (int it = 0; it < KITER; ++it) {
        const int no = it * 32;

        // ---- stream weight byte planes: non-temporal b128 loads ----
        i4v r0 = __builtin_nontemporal_load((const i4v*)(remRow + no + 0));
        i4v r1 = __builtin_nontemporal_load((const i4v*)(remRow + no + 4));
        i4v r2 = __builtin_nontemporal_load((const i4v*)(remRow + no + 8));
        i4v r3 = __builtin_nontemporal_load((const i4v*)(remRow + no + 12));
        i4v e0 = __builtin_nontemporal_load((const i4v*)(expRow + no + 0));
        i4v e1 = __builtin_nontemporal_load((const i4v*)(expRow + no + 4));
        i4v e2 = __builtin_nontemporal_load((const i4v*)(expRow + no + 8));
        i4v e3 = __builtin_nontemporal_load((const i4v*)(expRow + no + 12));

        // prefetch ~2 iterations ahead (speculative; dropped if past the end)
        __builtin_prefetch(remRow + no + 64, 0, 0);
        __builtin_prefetch(expRow + no + 64, 0, 0);

        // ---- decode via v_perm_b32: bf16 bits = (exp << 8) | rem ----
        union { unsigned int d[8]; v16bf v; } W;
        W.d[0] = pack2((unsigned)r0[0], (unsigned)r0[1], (unsigned)e0[0], (unsigned)e0[1]);
        W.d[1] = pack2((unsigned)r0[2], (unsigned)r0[3], (unsigned)e0[2], (unsigned)e0[3]);
        W.d[2] = pack2((unsigned)r1[0], (unsigned)r1[1], (unsigned)e1[0], (unsigned)e1[1]);
        W.d[3] = pack2((unsigned)r1[2], (unsigned)r1[3], (unsigned)e1[2], (unsigned)e1[3]);
        W.d[4] = pack2((unsigned)r2[0], (unsigned)r2[1], (unsigned)e2[0], (unsigned)e2[1]);
        W.d[5] = pack2((unsigned)r2[2], (unsigned)r2[3], (unsigned)e2[2], (unsigned)e2[3]);
        W.d[6] = pack2((unsigned)r3[0], (unsigned)r3[1], (unsigned)e3[0], (unsigned)e3[1]);
        W.d[7] = pack2((unsigned)r3[2], (unsigned)r3[3], (unsigned)e3[2], (unsigned)e3[3]);

        // ---- A operand: 16x32 bf16 x-tile (two b128 loads, L2-resident) ----
        union { u16x8 h[2]; v16bf v; } A;
        A.h[0] = *(const u16x8*)(xBase + no + half * 16);
        A.h[1] = *(const u16x8*)(xBase + no + half * 16 + 8);

        // D(16x16 f32) += A(16x32 bf16) x B(32x16 bf16)
        acc = __builtin_amdgcn_wmma_f32_16x16x32_bf16(
                  false, A.v, false, W.v, (short)0, acc, false, false);
    }

    // ---- cross-wave split-K reduction through LDS ----
    #pragma unroll
    for (int j = 0; j < 8; ++j) red[wave][lane][j] = acc[j];
    __syncthreads();

    if (tid < 16 * BATCH) {
        const int b = tid >> 4;     // batch row (D-tile comps 0..3, lanes 0..15)
        const int l = tid & 15;     // m offset within tile
        float sum = 0.f;
        #pragma unroll
        for (int w = 0; w < WAVES; ++w) sum += red[w][l][b];
        sum += __bfloat162float(*(const __hip_bfloat16*)(bias + m0 + l));
        *(__hip_bfloat16*)(out + (size_t)b * MDIM + m0 + l) = __float2bfloat16(sum);
    }
}

extern "C" void kernel_launch(void* const* d_in, const int* in_sizes, int n_in,
                              void* d_out, int out_size, void* d_ws, size_t ws_size,
                              hipStream_t stream) {
    (void)in_sizes; (void)n_in; (void)out_size; (void)d_ws; (void)ws_size;
    const unsigned short* x    = (const unsigned short*)d_in[0];
    const int*            rem  = (const int*)d_in[1];
    const int*            expb = (const int*)d_in[2];
    const unsigned short* bias = (const unsigned short*)d_in[3];
    unsigned short*       out  = (unsigned short*)d_out;

    dim3 grid(MDIM / 16);        // 512 tiles of 16 output rows
    dim3 block(WAVES * 32);      // 8 wave32 waves, split-K
    hipLaunchKernelGGL(huff_gemv_wmma, grid, block, 0, stream,
                       x, rem, expb, bias, out);
}
